// MemoryBlock_84748294685203
// MI455X (gfx1250) — matrix operand order repaired and checked by reading, not compiled, not run on hardware
//
#include <hip/hip_runtime.h>
#include <math.h>

// ---------------- problem constants ----------------
#define BATCH    16384
#define FEAT     512
#define NMEM     2000
#define NPAD     2048         // NMEM padded to 128 tiles of 16 -> branch-free GEMMs
#define NT1      128          // GEMM1 N-tiles (memories / 16)
#define THRESH   (1.0f / 2000.0f)
#define ENT_COEF 0.0002f

typedef __bf16 bf16_t;
typedef __attribute__((ext_vector_type(8)))  __bf16 v8bf;
typedef __attribute__((ext_vector_type(16))) __bf16 v16bf;
typedef __attribute__((ext_vector_type(8)))  float  v8f;

// =====================================================================
// Kernel 0: prep memory matrix (padded to NPAD rows).
//   m <  NMEM: memb[m][:] = bf16(memory[m][:]); memt[f][m] = bf16(...);
//              mn[m] = ||memory[m]||_2 (fp32).
//   m >= NMEM: memb row = 0, memt column = 0, mn[m] = 1.
// =====================================================================
__global__ __launch_bounds__(256) void prep_kernel(
    const float* __restrict__ mem,
    float* __restrict__ mn,
    bf16_t* __restrict__ memb,
    bf16_t* __restrict__ memt)
{
    __shared__ float red[256];
    const int m = blockIdx.x;
    const int t = threadIdx.x;

    if (m >= NMEM) {
        memb[(size_t)m * FEAT + t]         = (bf16_t)0.0f;
        memb[(size_t)m * FEAT + t + 256]   = (bf16_t)0.0f;
        memt[(size_t)t         * NPAD + m] = (bf16_t)0.0f;
        memt[(size_t)(t + 256) * NPAD + m] = (bf16_t)0.0f;
        if (t == 0) mn[m] = 1.0f;
        return;
    }

    float v0 = mem[(size_t)m * FEAT + t];
    float v1 = mem[(size_t)m * FEAT + t + 256];
    memb[(size_t)m * FEAT + t]         = (bf16_t)v0;
    memb[(size_t)m * FEAT + t + 256]   = (bf16_t)v1;
    memt[(size_t)t         * NPAD + m] = (bf16_t)v0;
    memt[(size_t)(t + 256) * NPAD + m] = (bf16_t)v1;

    red[t] = v0 * v0 + v1 * v1;
    __syncthreads();
    for (int s = 128; s > 0; s >>= 1) {
        if (t < s) red[t] += red[t + s];
        __syncthreads();
    }
    if (t == 0) mn[m] = sqrtf(red[0]);
}

// =====================================================================
// Kernel 1: fused strip kernel. One workgroup (256 thr = 8 waves) owns
// 16 batch rows. GEMM1 (bf16 WMMA, branch-free over 128 N-tiles) ->
// cosine scale into f32 LDS strip -> softmax/shrink/L1-renorm/entropy,
// writing w as bf16 into a second LDS strip -> GEMM2 (bf16 WMMA, A from
// LDS) -> global out. `w` never leaves the WGP.
// =====================================================================
__global__ __launch_bounds__(256) void memblock_kernel(
    const float*  __restrict__ x,
    const bf16_t* __restrict__ memb,   // [NPAD][FEAT] bf16, row major
    const bf16_t* __restrict__ memt,   // [FEAT][NPAD] bf16, transposed
    const float*  __restrict__ mn,     // [NPAD] fp32 row norms (1 for pads)
    float* __restrict__ out,           // [BATCH][FEAT]
    float* __restrict__ ent_out)       // scalar (pre-zeroed)
{
    extern __shared__ char smem[];
    float*  logits = (float*)smem;                                   // 16*2048 f32   (128 KB)
    bf16_t* wbf    = (bf16_t*)(smem + 16 * NPAD * sizeof(float));    // 16*2048 bf16  ( 64 KB)
    bf16_t* xs     = wbf + 16 * NPAD;                                // 16*512  bf16  ( 16 KB)
    float*  red    = (float*)(xs + 16 * FEAT);                       // 256 f32
    float*  zn     = red + 256;                                      // 16 f32
    float*  entAcc = zn + 16;                                        // 1 f32

    const int t    = threadIdx.x;
    const int wave = t >> 5;
    const int lane = t & 31;
    const int b0   = blockIdx.x * 16;

    // ---- stage x strip as bf16 into LDS (coalesced) ----
    for (int i = t; i < 16 * FEAT; i += 256) {
        int r = i >> 9, f = i & (FEAT - 1);
        xs[i] = (bf16_t)x[(size_t)(b0 + r) * FEAT + f];
    }
    // ---- fp32 row norms of x strip (16 threads per row) ----
    {
        int r = t >> 4, c = t & 15;
        float s = 0.f;
        for (int f = c; f < FEAT; f += 16) {
            float v = x[(size_t)(b0 + r) * FEAT + f];
            s += v * v;
        }
        red[t] = s;
    }
    if (t == 0) entAcc[0] = 0.f;
    __syncthreads();
    if (t < 16) {
        float s = 0.f;
        #pragma unroll
        for (int c = 0; c < 16; ++c) s += red[t * 16 + c];
        zn[t] = sqrtf(s);
    }
    __syncthreads();

    const int arow = lane & 15;            // A row / B column within tile
    const int koff = (lane >> 4) << 3;     // unused: A loads use boff (0 or 16) like B
    const int boff = (lane >> 4) << 4;     // B: K half 0 or 16

    // =================== GEMM1: logits = x @ memT ===================
    // 128 N-tiles of 16 memories; chunks of 4 tiles share the A operand.
    for (int ch = wave; ch < NT1 / 4; ch += 8) {
        const int ntb = ch * 4;
        v8f acc[4] = {v8f{}, v8f{}, v8f{}, v8f{}};

        const bf16_t* b0p = memb + (size_t)(ntb * 16 + arow) * FEAT + boff;
        __builtin_prefetch(b0p, 0, 1);   // warm this chunk's first line

        #pragma unroll 4
        for (int k0 = 0; k0 < FEAT; k0 += 32) {
            v8bf alo = *(const v8bf*)(xs + arow * FEAT + k0 + boff);
            v8bf ahi = *(const v8bf*)(xs + arow * FEAT + k0 + boff + 8);
            v16bf a = __builtin_shufflevector(alo, ahi,
                0, 1, 2, 3, 4, 5, 6, 7, 8, 9, 10, 11, 12, 13, 14, 15);
            #pragma unroll
            for (int j = 0; j < 4; ++j) {
                v16bf b = *(const v16bf*)(b0p + (size_t)(j * 16) * FEAT + k0);
                acc[j] = __builtin_amdgcn_wmma_f32_16x16x32_bf16(
                    false, a, false, b, (short)0, acc[j], false, false);
            }
        }
        // cosine scale + spill strip to LDS (branch-free; pads harmless)
        #pragma unroll
        for (int j = 0; j < 4; ++j) {
            const int mcol = (ntb + j) * 16 + arow;
            const float mnv = mn[mcol];
            #pragma unroll
            for (int r = 0; r < 8; ++r) {
                const int orow = (lane < 16) ? r : (8 + r);
                const float denom = fmaxf(zn[orow] * mnv, 1e-8f);
                logits[orow * NPAD + mcol] = acc[j][r] / denom;
            }
        }
    }
    __syncthreads();

    // ====== softmax -> hard-shrink -> L1 renorm -> entropy (per row) ======
    for (int r = wave * 2; r < wave * 2 + 2; ++r) {
        float* row = logits + r * NPAD;

        float mx = -3.4e38f;
        for (int c = lane; c < NMEM; c += 32) mx = fmaxf(mx, row[c]);
        for (int o = 16; o > 0; o >>= 1) mx = fmaxf(mx, __shfl_xor(mx, o));

        float s = 0.f;
        for (int c = lane; c < NMEM; c += 32) {
            float e = __expf(row[c] - mx);
            row[c] = e;
            s += e;
        }
        for (int o = 16; o > 0; o >>= 1) s += __shfl_xor(s, o);
        const float inv = 1.f / s;

        float s1 = 0.f;
        for (int c = lane; c < NMEM; c += 32) {
            float p = row[c] * inv;
            float d = p - THRESH;
            float wv = fmaxf(d, 0.f) * p / (fabsf(d) + 1e-12f);
            row[c] = wv;
            s1 += wv;
        }
        for (int o = 16; o > 0; o >>= 1) s1 += __shfl_xor(s1, o);
        const float invL1 = 1.f / fmaxf(s1, 1e-12f);

        float ent = 0.f;
        for (int c = lane; c < NMEM; c += 32) {
            float wn = row[c] * invL1;
            wbf[r * NPAD + c] = (bf16_t)wn;           // GEMM2 A operand
            ent += -wn * __logf(wn + 1e-12f);
        }
        for (int o = 16; o > 0; o >>= 1) ent += __shfl_xor(ent, o);
        if (lane == 0) atomicAdd(entAcc, ent);
    }
    // zero K-padding columns of the bf16 w strip: 16 rows x 48 cols
    for (int i = t; i < 16 * (NPAD - NMEM); i += 256) {
        int r = i / (NPAD - NMEM), c = i % (NPAD - NMEM);
        wbf[r * NPAD + NMEM + c] = (bf16_t)0.0f;
    }
    __syncthreads();
    if (t == 0) atomicAdd(ent_out, ENT_COEF * entAcc[0]);

    // =================== GEMM2: out = w @ memory ===================
    // 32 F-tiles of 16; wave w owns tiles 4w..4w+3, K = NPAD over LDS w.
    {
        v8f acc[4] = {v8f{}, v8f{}, v8f{}, v8f{}};
        const bf16_t* b0p = memt + (size_t)(wave * 64 + arow) * NPAD + boff;
        __builtin_prefetch(b0p, 0, 1);

        #pragma unroll 4
        for (int m0 = 0; m0 < NPAD; m0 += 32) {
            v8bf alo = *(const v8bf*)(wbf + arow * NPAD + m0 + boff);
            v8bf ahi = *(const v8bf*)(wbf + arow * NPAD + m0 + boff + 8);
            v16bf a = __builtin_shufflevector(alo, ahi,
                0, 1, 2, 3, 4, 5, 6, 7, 8, 9, 10, 11, 12, 13, 14, 15);
            #pragma unroll
            for (int j = 0; j < 4; ++j) {
                v16bf b = *(const v16bf*)(b0p + (size_t)(j * 16) * NPAD + m0);
                acc[j] = __builtin_amdgcn_wmma_f32_16x16x32_bf16(
                    false, a, false, b, (short)0, acc[j], false, false);
            }
        }
        #pragma unroll
        for (int j = 0; j < 4; ++j) {
            const int f = (wave * 4 + j) * 16 + arow;
            #pragma unroll
            for (int r = 0; r < 8; ++r) {
                const int orow = (lane < 16) ? r : (8 + r);
                out[(size_t)(b0 + orow) * FEAT + f] = acc[j][r];
            }
        }
    }
}

// =====================================================================
// Host launcher
// =====================================================================
extern "C" void kernel_launch(void* const* d_in, const int* in_sizes, int n_in,
                              void* d_out, int out_size, void* d_ws, size_t ws_size,
                              hipStream_t stream)
{
    const float* x   = (const float*)d_in[0];   // [BATCH][FEAT]
    const float* mem = (const float*)d_in[1];   // [NMEM][FEAT]
    float* out  = (float*)d_out;                         // [BATCH][FEAT]
    float* entp = out + (size_t)BATCH * FEAT;            // trailing scalar

    // workspace layout (16B-aligned):
    //   mn   : NPAD f32            ->   8192 B
    //   memb : NPAD*FEAT bf16      -> 2 MiB
    //   memt : FEAT*NPAD bf16      -> 2 MiB
    char* ws = (char*)d_ws;
    float*  mn   = (float*)ws;
    bf16_t* memb = (bf16_t*)(ws + 8192);
    bf16_t* memt = (bf16_t*)(ws + 8192 + (size_t)NPAD * FEAT * sizeof(bf16_t));

    hipMemsetAsync(entp, 0, sizeof(float), stream);

    prep_kernel<<<NPAD, 256, 0, stream>>>(mem, mn, memb, memt);

    const size_t smem = (size_t)16 * NPAD * sizeof(float)    // f32 logit strip
                      + (size_t)16 * NPAD * sizeof(bf16_t)   // bf16 w strip
                      + (size_t)16 * FEAT * sizeof(bf16_t)   // x strip (bf16)
                      + (256 + 16 + 1) * sizeof(float) + 12; // red + zn + entAcc
    hipFuncSetAttribute((const void*)memblock_kernel,
                        hipFuncAttributeMaxDynamicSharedMemorySize, (int)smem);

    memblock_kernel<<<BATCH / 16, 256, smem, stream>>>(x, memb, memt, mn, out, entp);
}
